// MFNN_77455440216336
// MI455X (gfx1250) — matrix operand order repaired and checked by reading, not compiled, not run on hardware
//
#include <hip/hip_runtime.h>

typedef __bf16 bf16;
typedef __attribute__((ext_vector_type(8)))  __bf16 v8bf;
typedef __attribute__((ext_vector_type(16))) __bf16 v16bf;
typedef __attribute__((ext_vector_type(8)))  float  v8f;

constexpr int kN  = 100000;   // nodes
constexpr int kF  = 500;      // raw feature dim
constexpr int kFP = 512;      // padded feature dim
constexpr int kM  = 512;      // spectral dim
constexpr int kH  = 128;      // hidden
constexpr int kC  = 40;       // classes
constexpr int kSplits = 125;  // split-K for u^T h
constexpr int kChunk  = 800;  // kSplits*kChunk == kN, kChunk % 32 == 0

// ---------- bf16 helpers (bit-level, RNE round; avoids relying on __bf16 arithmetic) ----------
static __device__ __forceinline__ bf16 f2bf(float f) {
  unsigned u = __builtin_bit_cast(unsigned, f);
  u += 0x7FFFu + ((u >> 16) & 1u);
  unsigned short h = (unsigned short)(u >> 16);
  return __builtin_bit_cast(bf16, h);
}
static __device__ __forceinline__ float bf2f(bf16 b) {
  unsigned short h = __builtin_bit_cast(unsigned short, b);
  unsigned u = ((unsigned)h) << 16;
  return __builtin_bit_cast(float, u);
}

// A fragment: lane half holds K {kb..kb+7} and {kb+16..kb+23}  (16-bit A 16x32 layout)
static __device__ __forceinline__ v16bf load_a_frag(const bf16* p) {
  v8bf lo = *(const v8bf*)p;
  v8bf hi = *(const v8bf*)(p + 16);
  v16bf r;
#pragma unroll
  for (int i = 0; i < 8; ++i) { r[i] = lo[i]; r[i + 8] = hi[i]; }
  return r;
}
// B fragment: lane half holds K {kb..kb+7} and {kb+16..kb+23} (matches A layout)
static __device__ __forceinline__ v16bf load_b_frag(const bf16* p) {
  v8bf lo = *(const v8bf*)p;
  v8bf hi = *(const v8bf*)(p + 16);
  v16bf r;
#pragma unroll
  for (int i = 0; i < 8; ++i) { r[i] = lo[i]; r[i + 8] = hi[i]; }
  return r;
}

enum { EPI_PARTIAL_F32 = 0, EPI_ROW_BF16_RELU_BIAS = 1, EPI_TRANS_BF16_BIAS = 2 };

// Generic bf16 WMMA GEMM: C[M,128] = A[M,K] * B[K,128], with Bt = B^T stored row-major [128,K].
// Block = 8 waves; wave (wr,wc) computes rows [wr*32,+32) x cols [bY*64+wc*32,+32).
// blockIdx.z = split-K slice (A/Bt advanced by z*K columns, partial out offset z*partialStride).
__global__ __launch_bounds__(256) void gemm_bf16_kernel(
    const bf16* __restrict__ A, int lda,
    const bf16* __restrict__ Bt, int ldb,
    int M, int K,
    void* __restrict__ out, int ldo,
    const float* __restrict__ bias,
    int epi, long partialStride)
{
  const int tid   = threadIdx.x;
  const int wave  = tid >> 5;
  const int lane  = tid & 31;
  const int lr    = lane & 15;
  const int lhalf = lane >> 4;
  const int wr    = wave & 3;
  const int wc    = wave >> 2;

  const long row0 = (long)blockIdx.x * 128 + wr * 32;
  const int  col0 = blockIdx.y * 64 + wc * 32;
  const long zoff = (long)blockIdx.z * K;

  long ra0 = row0 + lr;      if (ra0 > M - 1) ra0 = M - 1;
  long ra1 = row0 + 16 + lr; if (ra1 > M - 1) ra1 = M - 1;

  const bf16* pa0 = A + ra0 * (long)lda + zoff + lhalf * 8;
  const bf16* pa1 = A + ra1 * (long)lda + zoff + lhalf * 8;
  const bf16* pb0 = Bt + (long)(col0 + lr) * ldb + zoff + lhalf * 8;
  const bf16* pb1 = Bt + (long)(col0 + 16 + lr) * ldb + zoff + lhalf * 8;

  v8f acc00 = {0.f,0.f,0.f,0.f,0.f,0.f,0.f,0.f};
  v8f acc01 = acc00, acc10 = acc00, acc11 = acc00;

  for (int kk = 0; kk < K; kk += 32) {
    __builtin_prefetch(pa0 + kk + 64, 0, 1);
    __builtin_prefetch(pb0 + kk + 64, 0, 1);
    v16bf a0 = load_a_frag(pa0 + kk);
    v16bf a1 = load_a_frag(pa1 + kk);
    v16bf b0 = load_b_frag(pb0 + kk);
    v16bf b1 = load_b_frag(pb1 + kk);
    acc00 = __builtin_amdgcn_wmma_f32_16x16x32_bf16(false, a0, false, b0, (short)0, acc00, false, false);
    acc01 = __builtin_amdgcn_wmma_f32_16x16x32_bf16(false, a0, false, b1, (short)0, acc01, false, false);
    acc10 = __builtin_amdgcn_wmma_f32_16x16x32_bf16(false, a1, false, b0, (short)0, acc10, false, false);
    acc11 = __builtin_amdgcn_wmma_f32_16x16x32_bf16(false, a1, false, b1, (short)0, acc11, false, false);
  }

  v8f accs[2][2] = {{acc00, acc01}, {acc10, acc11}};
#pragma unroll
  for (int mi = 0; mi < 2; ++mi) {
#pragma unroll
    for (int ni = 0; ni < 2; ++ni) {
      v8f acc = accs[mi][ni];
      const long rbase = row0 + mi * 16 + lhalf * 8;  // C/D: M = i + 8*lhalf
      const int  col   = col0 + ni * 16 + lr;          // C/D: N = lr
#pragma unroll
      for (int i = 0; i < 8; ++i) {
        const long r = rbase + i;
        if (r < M) {
          float v = acc[i];
          if (epi == EPI_PARTIAL_F32) {
            float* o = (float*)out + (long)blockIdx.z * partialStride;
            o[r * (long)ldo + col] = v;
          } else if (epi == EPI_ROW_BF16_RELU_BIAS) {
            v += bias[col];
            v = v > 0.f ? v : 0.f;
            ((bf16*)out)[r * (long)ldo + col] = f2bf(v);
          } else {  // EPI_TRANS_BF16_BIAS (bias may be null)
            if (bias) v += bias[col];
            ((bf16*)out)[(long)col * (long)ldo + r] = f2bf(v);
          }
        }
      }
    }
  }
}

// wt[c*ldt + k] = (k < krows) ? w[k*128 + c] : 0   (build B^T in bf16, zero-padded K)
__global__ void transpose_pad_w_kernel(const float* __restrict__ w, bf16* __restrict__ wt,
                                       int krows, int ldt) {
  int idx = blockIdx.x * blockDim.x + threadIdx.x;
  int total = kH * ldt;
  if (idx >= total) return;
  int c = idx / ldt, k = idx - c * ldt;
  wt[idx] = (k < krows) ? f2bf(w[(long)k * kH + c]) : f2bf(0.f);
}

// x [N,500] fp32 -> xb [N,512] bf16 (zero pad)
__global__ void cvt_x_kernel(const float* __restrict__ x, bf16* __restrict__ xb) {
  long idx = (long)blockIdx.x * blockDim.x + threadIdx.x;
  if (idx >= (long)kN * kFP) return;
  long n = idx >> 9; int k = (int)(idx & 511);
  xb[idx] = (k < kF) ? f2bf(x[n * kF + k]) : f2bf(0.f);
}

// u [N,512] fp32 -> ub [N,512] bf16 (row major) and ut [512,N] bf16 (transposed, via LDS tile)
__global__ void cvt_u_kernel(const float* __restrict__ u, bf16* __restrict__ ub,
                             bf16* __restrict__ ut) {
  __shared__ float tile[32][33];
  int mBase = blockIdx.x * 32;
  int nBase = blockIdx.y * 32;
  int tx = threadIdx.x, ty = threadIdx.y;
#pragma unroll
  for (int r = 0; r < 4; ++r) {
    int n = nBase + ty + r * 8;
    float v = u[(long)n * kM + mBase + tx];
    ub[(long)n * kM + mBase + tx] = f2bf(v);
    tile[ty + r * 8][tx] = v;
  }
  __syncthreads();
#pragma unroll
  for (int r = 0; r < 4; ++r) {
    int mL = ty + r * 8;
    int nL = tx;
    ut[(long)(mBase + mL) * kN + nBase + nL] = f2bf(tile[nL][mL]);
  }
}

// eig[m] = eig_b + eig_w[0] + sum_{k=1..128} cos(pi*e[m]*k) * eig_w[k]
__global__ void eig_kernel(const float* __restrict__ e, const float* __restrict__ eig_w,
                           const float* __restrict__ eig_b, float* __restrict__ eig) {
  int m = blockIdx.x * blockDim.x + threadIdx.x;
  if (m >= kM) return;
  float ev = e[m];
  float acc = eig_w[0] + eig_b[0];
  for (int k = 1; k <= kH; ++k)
    acc += cosf(3.14159265358979323846f * ev * (float)k) * eig_w[k];
  eig[m] = acc;
}

// vt[c*512 + m] = bf16( eig[m] * sum_z partial[z][m][c] )   (V^T operand for r = u * V)
__global__ void reduce_utx_kernel(const float* __restrict__ partial, const float* __restrict__ eig,
                                  bf16* __restrict__ vt) {
  int idx = blockIdx.x * blockDim.x + threadIdx.x;
  if (idx >= kM * kH) return;
  int m = idx >> 7, c = idx & 127;
  float s = 0.f;
  for (int z = 0; z < kSplits; ++z) s += partial[((long)z * kM + m) * kH + c];
  vt[(long)c * kM + m] = f2bf(eig[m] * s);
}

// s = (w0 + fW0*w1) * h + (tail*w1) * r ; LayerNorm ; ReLU  (all in transposed [128,N] layout)
__global__ __launch_bounds__(256) void layer_kernel(
    const bf16* __restrict__ hin, const bf16* __restrict__ rt, bf16* __restrict__ hout,
    const float* __restrict__ lw, const float* __restrict__ gamma,
    const float* __restrict__ beta, const float* __restrict__ fW) {
  __shared__ float sa[kH], sb[kH], sg[kH], so[kH];
  int t = threadIdx.x;
  if (t < kH) {
    float fW0 = fW[0], tail = 0.f;
    for (int k = 1; k <= 10; ++k) tail += fW[k];
    float w0 = lw[t], w1 = lw[kH + t];
    sa[t] = w0 + fW0 * w1;
    sb[t] = tail * w1;
    sg[t] = gamma[t];
    so[t] = beta[t];
  }
  __syncthreads();
  long n = (long)blockIdx.x * blockDim.x + t;
  if (n >= kN) return;
  float sum = 0.f, sq = 0.f;
  for (int c = 0; c < kH; ++c) {
    float s = sa[c] * bf2f(hin[(long)c * kN + n]) + sb[c] * bf2f(rt[(long)c * kN + n]);
    sum += s; sq += s * s;
  }
  float mu  = sum * (1.f / kH);
  float var = sq * (1.f / kH) - mu * mu;
  float inv = rsqrtf(var + 1e-5f);
  for (int c = 0; c < kH; ++c) {
    float s = sa[c] * bf2f(hin[(long)c * kN + n]) + sb[c] * bf2f(rt[(long)c * kN + n]);
    float o = (s - mu) * inv * sg[c] + so[c];
    hout[(long)c * kN + n] = f2bf(o > 0.f ? o : 0.f);
  }
}

// attention combine + lin3 + log_softmax
__global__ __launch_bounds__(128) void final_kernel(
    const bf16* __restrict__ h0t, const bf16* __restrict__ h1t, const bf16* __restrict__ h2t,
    const float* __restrict__ att0, const float* __restrict__ att1, const float* __restrict__ att2,
    const float* __restrict__ attv, const float* __restrict__ w3, const float* __restrict__ b3,
    float* __restrict__ out) {
  __shared__ float sA0[kH], sA1[kH], sA2[kH], sV[9], sW[kH * kC], sB[kC];
  for (int i = threadIdx.x; i < kH; i += blockDim.x) { sA0[i]=att0[i]; sA1[i]=att1[i]; sA2[i]=att2[i]; }
  for (int i = threadIdx.x; i < 9; i += blockDim.x) sV[i] = attv[i];
  for (int i = threadIdx.x; i < kH * kC; i += blockDim.x) sW[i] = w3[i];
  for (int i = threadIdx.x; i < kC; i += blockDim.x) sB[i] = b3[i];
  __syncthreads();
  long n = (long)blockIdx.x * blockDim.x + threadIdx.x;
  if (n >= kN) return;
  float d0 = 0.f, d1 = 0.f, d2 = 0.f;
  for (int c = 0; c < kH; ++c) {
    d0 += bf2f(h0t[(long)c * kN + n]) * sA0[c];
    d1 += bf2f(h1t[(long)c * kN + n]) * sA1[c];
    d2 += bf2f(h2t[(long)c * kN + n]) * sA2[c];
  }
  float l0 = 1.f / (1.f + expf(-d0));
  float l1 = 1.f / (1.f + expf(-d1));
  float l2 = 1.f / (1.f + expf(-d2));
  float g0 = l0 * sV[0] + l1 * sV[3] + l2 * sV[6];
  float g1 = l0 * sV[1] + l1 * sV[4] + l2 * sV[7];
  float g2 = l0 * sV[2] + l1 * sV[5] + l2 * sV[8];
  float mx = fmaxf(g0, fmaxf(g1, g2));
  float x0 = expf(g0 - mx), x1 = expf(g1 - mx), x2 = expf(g2 - mx);
  float inv = 1.f / (x0 + x1 + x2);
  float a0 = x0 * inv, a1 = x1 * inv, a2 = x2 * inv;
  float hc[kH];
  for (int c = 0; c < kH; ++c)
    hc[c] = a0 * bf2f(h0t[(long)c * kN + n]) + a1 * bf2f(h1t[(long)c * kN + n])
          + a2 * bf2f(h2t[(long)c * kN + n]);
  float tv[kC];
  float tmax = -1e30f;
  for (int j = 0; j < kC; ++j) {
    float acc = sB[j];
    for (int c = 0; c < kH; ++c) acc += hc[c] * sW[c * kC + j];
    tv[j] = acc;
    tmax = fmaxf(tmax, acc);
  }
  float se = 0.f;
  for (int j = 0; j < kC; ++j) se += expf(tv[j] - tmax);
  float lse = tmax + logf(se);
  for (int j = 0; j < kC; ++j) out[n * kC + j] = tv[j] - lse;
}

extern "C" void kernel_launch(void* const* d_in, const int* in_sizes, int n_in,
                              void* d_out, int out_size, void* d_ws, size_t ws_size,
                              hipStream_t stream) {
  (void)in_sizes; (void)n_in; (void)out_size; (void)ws_size;
  const float* x        = (const float*)d_in[0];
  const float* e0       = (const float*)d_in[1];
  const float* e1       = (const float*)d_in[2];
  const float* u0       = (const float*)d_in[3];
  const float* u1       = (const float*)d_in[4];
  const float* fe_w1    = (const float*)d_in[5];
  const float* fe_b1    = (const float*)d_in[6];
  const float* fe_w2    = (const float*)d_in[7];
  const float* fe_b2    = (const float*)d_in[8];
  const float* eig_w    = (const float*)d_in[9];
  const float* eig_b    = (const float*)d_in[10];
  const float* layer_w  = (const float*)d_in[11];
  const float* ln_gamma = (const float*)d_in[12];
  const float* ln_beta  = (const float*)d_in[13];
  const float* fW       = (const float*)d_in[14];
  const float* att0     = (const float*)d_in[15];
  const float* att1     = (const float*)d_in[16];
  const float* att2     = (const float*)d_in[17];
  const float* att_vec  = (const float*)d_in[18];
  const float* lin3_w   = (const float*)d_in[19];
  const float* lin3_b   = (const float*)d_in[20];

  char* ws = (char*)d_ws;
  size_t off = 0;
  auto take = [&](size_t bytes) -> void* {
    void* p = ws + off;
    off += (bytes + 255) & ~(size_t)255;
    return p;
  };
  // ~366 MB total workspace
  bf16* xb   = (bf16*)take((size_t)kN * kFP * 2);  // xb (feat enc), later reused as ub
  bf16* ub   = xb;                                  // alias: xb dead after G1
  bf16* ut   = (bf16*)take((size_t)kM * kN * 2);
  bf16* t1   = (bf16*)take((size_t)kN * kH * 2);   // t1 (feat enc), later reused as rt
  bf16* rt   = t1;                                  // alias: t1 dead after G2
  bf16* hft  = (bf16*)take((size_t)kH * kN * 2);   // result[0]^T
  bf16* h1t  = (bf16*)take((size_t)kH * kN * 2);   // result[1]^T
  bf16* h2t  = (bf16*)take((size_t)kH * kN * 2);   // result[2]^T
  bf16* htm  = (bf16*)take((size_t)kH * kN * 2);   // intermediate layer output
  bf16* vt   = (bf16*)take((size_t)kH * kM * 2);   // (eig*utx)^T
  bf16* w1t  = (bf16*)take((size_t)kH * kFP * 2);
  bf16* w2t  = (bf16*)take((size_t)kH * kH * 2);
  float* eig     = (float*)take((size_t)kM * 4);
  float* partial = (float*)take((size_t)kSplits * kM * kH * 4);

  // Weight prep + x conversion
  transpose_pad_w_kernel<<<(kH * kFP + 255) / 256, 256, 0, stream>>>(fe_w1, w1t, kF, kFP);
  transpose_pad_w_kernel<<<(kH * kH + 255) / 256, 256, 0, stream>>>(fe_w2, w2t, kH, kH);
  cvt_x_kernel<<<(int)(((long)kN * kFP + 255) / 256), 256, 0, stream>>>(x, xb);

  dim3 gBig((kN + 127) / 128, 2, 1);
  // G1: t1 = relu(x @ W1 + b1)        [N,512]x[512,128] -> bf16 row-major
  gemm_bf16_kernel<<<gBig, 256, 0, stream>>>(xb, kFP, w1t, kFP, kN, kFP, t1, kH, fe_b1,
                                             EPI_ROW_BF16_RELU_BIAS, 0);
  // G2: h = t1 @ W2 + b2              [N,128]x[128,128] -> bf16 transposed hft[128,N]
  gemm_bf16_kernel<<<gBig, 256, 0, stream>>>(t1, kH, w2t, kH, kN, kH, hft, kN, fe_b2,
                                             EPI_TRANS_BF16_BIAS, 0);

  const float* es[2] = {e0, e1};
  const float* us[2] = {u0, u1};
  bf16* hstart[2] = {hft, h1t};   // h carries over between branches
  bf16* hend[2]   = {h1t, h2t};
  for (int br = 0; br < 2; ++br) {
    eig_kernel<<<2, 256, 0, stream>>>(es[br], eig_w, eig_b, eig);
    cvt_u_kernel<<<dim3(kM / 32, kN / 32), dim3(32, 8), 0, stream>>>(us[br], ub, ut);
    bf16* hin = hstart[br];
    for (int L = 0; L < 2; ++L) {
      bf16* hout = (L == 0) ? htm : hend[br];
      // G3 (split-K): partial[z] = ut[:, z*800:(z+1)*800] @ h[z*800:(z+1)*800, :]
      gemm_bf16_kernel<<<dim3(kM / 128, 2, kSplits), 256, 0, stream>>>(
          ut, kN, hin, kN, kM, kChunk, partial, kH, nullptr,
          EPI_PARTIAL_F32, (long)kM * kH);
      reduce_utx_kernel<<<(kM * kH + 255) / 256, 256, 0, stream>>>(partial, eig, vt);
      // G4: r = u @ (eig*utx)          [N,512]x[512,128] -> bf16 transposed rt[128,N]
      gemm_bf16_kernel<<<gBig, 256, 0, stream>>>(ub, kFP, vt, kM, kN, kM, rt, kN, nullptr,
                                                 EPI_TRANS_BF16_BIAS, 0);
      layer_kernel<<<(kN + 255) / 256, 256, 0, stream>>>(
          hin, rt, hout, layer_w + (long)L * 2 * kH,
          ln_gamma + (long)L * kH, ln_beta + (long)L * kH, fW);
      hin = hout;
    }
  }

  final_kernel<<<(kN + 127) / 128, 128, 0, stream>>>(hft, h1t, h2t, att0, att1, att2,
                                                     att_vec, lin3_w, lin3_b, (float*)d_out);
}
